// Attention_49924699848781
// MI455X (gfx1250) — compile-verified, matrix operand order checked by reading, not run on hardware
//
#include <hip/hip_runtime.h>
#include <hip/hip_bf16.h>

// Problem constants (from reference setup_inputs)
constexpr int Bb = 8;
constexpr int Ss = 2048;
constexpr int Tt = 2048;
constexpr int Dd = 1024;
constexpr int Oo = 1024;
constexpr float NEG_BIG = -1e16f;

typedef __bf16 bf16_t;
typedef __attribute__((ext_vector_type(16))) __bf16 v16bf;
typedef __attribute__((ext_vector_type(8)))  __bf16 v8bf;
typedef __attribute__((ext_vector_type(4)))  __bf16 v4bf;
typedef __attribute__((ext_vector_type(8)))  float  v8f;
typedef __attribute__((ext_vector_type(4)))  float  v4f;

// fp32 -> bf16 round-to-nearest-even
__device__ __forceinline__ bf16_t f2bf(float f) {
    unsigned u = __builtin_bit_cast(unsigned, f);
    u += 0x7FFFu + ((u >> 16) & 1u);
    unsigned short h = (unsigned short)(u >> 16);
    return __builtin_bit_cast(bf16_t, h);
}

// LDS tile: 128 rows x 32 K (bf16), padded to 40 elems (80B rows, 16B aligned)
#define LPAD 40

// ---- gfx1250 async global->LDS copy (ASYNCcnt) -----------------------------
__device__ __forceinline__ void async_cp_b128(const bf16_t* lds_p, const bf16_t* gp) {
    unsigned loff = (unsigned)(unsigned long long)(const void*)lds_p; // low 32b of generic LDS ptr
    asm volatile("global_load_async_to_lds_b128 %0, %1, off"
                 :: "v"(loff), "v"(gp)
                 : "memory");
}
__device__ __forceinline__ void wait_async0() {
    asm volatile("s_wait_asynccnt 0x0" ::: "memory");
}

// Stage a 128x32 bf16 tile (row-major, row stride ld elems) into LDS, async.
// 256 threads * 2 rounds * 16B = 8KB.
__device__ __forceinline__ void stage_async_rowmajor(bf16_t (*sT)[LPAD],
                                                     const bf16_t* __restrict__ g, int ld) {
    int seg = threadIdx.x;
#pragma unroll
    for (int it = 0; it < 2; ++it, seg += 256) {
        int row = seg >> 2;          // 0..127
        int kc  = (seg & 3) << 3;    // 0,8,16,24 (bf16 elems)
        async_cp_b128(&sT[row][kc], g + (size_t)row * ld + kc);
    }
}

// Synchronous transposed stage: sT[n][kk] = g[kk*ld + n] (bf16 source).
__device__ __forceinline__ void stage_colmajor_bf(bf16_t (*sT)[LPAD],
                                                  const bf16_t* __restrict__ g, int ld) {
    int kk = threadIdx.x >> 3;           // 0..31
    int n0 = (threadIdx.x & 7) << 4;     // 0..112 step 16
    const bf16_t* gp = g + (size_t)kk * ld + n0;
    v8bf a = *(const v8bf*)gp;
    v8bf b = *(const v8bf*)(gp + 8);
#pragma unroll
    for (int i = 0; i < 8; ++i) {
        sT[n0 + i][kk]     = a[i];
        sT[n0 + 8 + i][kk] = b[i];
    }
}

// Load one 16x32 bf16 fragment for v_wmma_f32_16x16x32_bf16.
// lanes 0-15: K 0..7 / 16..23 ; lanes 16-31: K 8..15 / 24..31.
__device__ __forceinline__ v16bf frag_ld(const bf16_t* p) {
    v8bf lo = *(const v8bf*)p;
    v8bf hi = *(const v8bf*)(p + 16);
    v16bf r;
#pragma unroll
    for (int i = 0; i < 8; ++i) { r[i] = lo[i]; r[i + 8] = hi[i]; }
    return r;
}

// ---------------------------------------------------------------------------
// Kernel 0: fp32 -> bf16 bulk convert (src/tgt/W into workspace)
// ---------------------------------------------------------------------------
__global__ __launch_bounds__(256) void k_cvt(const float* __restrict__ in,
                                             bf16_t* __restrict__ out, int n4) {
    int i = blockIdx.x * 256 + threadIdx.x;
    if (i >= n4) return;
    v4f f = *(const v4f*)(in + 4 * (size_t)i);
    v4bf h;
    h[0] = f2bf(f[0]); h[1] = f2bf(f[1]); h[2] = f2bf(f[2]); h[3] = f2bf(f[3]);
    *(v4bf*)(out + 4 * (size_t)i) = h;
}

// ---------------------------------------------------------------------------
// Kernel 1: scores[b,t,s] = dot(tgt[b,t,:], src[b,s,:]) + mask[t,b,s]*NEG_BIG
// Writes raw masked logits into the weight-output region (softmaxed in place).
// A/B staged via async-to-LDS, double buffered.
// ---------------------------------------------------------------------------
__global__ __launch_bounds__(256) void k_scores(const bf16_t* __restrict__ srch,
                                                const bf16_t* __restrict__ tgth,
                                                const int*    __restrict__ mask,
                                                float*        __restrict__ wout) {
    __shared__ __align__(16) bf16_t sA[2][128][LPAD];
    __shared__ __align__(16) bf16_t sB[2][128][LPAD];
    const int b     = blockIdx.z;
    const int tBase = blockIdx.y * 128;
    const int sBase = blockIdx.x * 128;
    const bf16_t* Ag = tgth + (size_t)b * Tt * Dd + (size_t)tBase * Dd;
    const bf16_t* Bg = srch + (size_t)b * Ss * Dd + (size_t)sBase * Dd;

    const int lane = threadIdx.x & 31;
    const int wave = threadIdx.x >> 5;
    const int wm = wave & 3, wn = wave >> 2;
    const int mr = lane & 15;
    const int kg = (lane >> 4) << 3;

    v8f zero = {0.f, 0.f, 0.f, 0.f, 0.f, 0.f, 0.f, 0.f};
    v8f acc[2][4];
#pragma unroll
    for (int i = 0; i < 2; ++i)
#pragma unroll
        for (int j = 0; j < 4; ++j) acc[i][j] = zero;

    stage_async_rowmajor(sA[0], Ag, Dd);
    stage_async_rowmajor(sB[0], Bg, Dd);
    wait_async0();
    __syncthreads();

    int p = 0;
    for (int k0 = 0; k0 < Dd; k0 += 32) {
        int kn = k0 + 32;
        if (kn < Dd) {
            stage_async_rowmajor(sA[p ^ 1], Ag + kn, Dd);
            stage_async_rowmajor(sB[p ^ 1], Bg + kn, Dd);
        }
        v16bf af[2], bfm[4];
#pragma unroll
        for (int i = 0; i < 2; ++i) af[i] = frag_ld(&sA[p][wm * 32 + i * 16 + mr][kg]);
#pragma unroll
        for (int j = 0; j < 4; ++j) bfm[j] = frag_ld(&sB[p][wn * 64 + j * 16 + mr][kg]);
#pragma unroll
        for (int i = 0; i < 2; ++i)
#pragma unroll
            for (int j = 0; j < 4; ++j)
                acc[i][j] = __builtin_amdgcn_wmma_f32_16x16x32_bf16(
                    false, af[i], false, bfm[j], (short)0, acc[i][j], false, false);
        wait_async0();
        __syncthreads();
        p ^= 1;
    }

    float* wb = wout + (size_t)b * Tt * Ss;
    const int half = lane >> 4;
    const int col  = lane & 15;
#pragma unroll
    for (int i = 0; i < 2; ++i) {
        int mb = tBase + wm * 32 + i * 16 + half * 8;
#pragma unroll
        for (int j = 0; j < 4; ++j) {
            int n = sBase + wn * 64 + j * 16 + col;
#pragma unroll
            for (int r = 0; r < 8; ++r) {
                int t = mb + r;
                float m = (float)mask[(size_t)t * (Bb * Ss) + (size_t)b * Ss + n];
                wb[(size_t)t * Ss + n] = acc[i][j][r] + m * NEG_BIG;
            }
        }
    }
}

// ---------------------------------------------------------------------------
// Kernel 2: in-place softmax over T (axis=1); also emits bf16 copy for GEMM 2.
// ---------------------------------------------------------------------------
__global__ __launch_bounds__(256) void k_softmax(float* __restrict__ w,
                                                 bf16_t* __restrict__ wh) {
    int idx = blockIdx.x * 256 + threadIdx.x;  // b*S + s
    int b = idx >> 11;
    int s = idx & (Ss - 1);
    size_t base = (size_t)b * Tt * Ss + s;
    float*  colp = w  + base;
    bf16_t* colh = wh + base;
    float mx = -3.4e38f;
    for (int t = 0; t < Tt; ++t) mx = fmaxf(mx, colp[(size_t)t * Ss]);
    float sum = 0.f;
    for (int t = 0; t < Tt; ++t) sum += __expf(colp[(size_t)t * Ss] - mx);
    float inv = 1.f / sum;
    for (int t = 0; t < Tt; ++t) {
        float v = __expf(colp[(size_t)t * Ss] - mx) * inv;
        colp[(size_t)t * Ss] = v;
        colh[(size_t)t * Ss] = f2bf(v);
    }
}

// ---------------------------------------------------------------------------
// Kernel 3: weight_sum[b,t,d] = sum_s weight[b,t,s] * src[b,s,d]  (bf16 out)
// A async, B = src^T staged synchronously (transpose), double buffered.
// ---------------------------------------------------------------------------
__global__ __launch_bounds__(256) void k_wsum(const bf16_t* __restrict__ wh,
                                              const bf16_t* __restrict__ srch,
                                              bf16_t*       __restrict__ wsumh) {
    __shared__ __align__(16) bf16_t sA[2][128][LPAD];
    __shared__ __align__(16) bf16_t sB[2][128][LPAD];
    const int b     = blockIdx.z;
    const int tBase = blockIdx.y * 128;
    const int dBase = blockIdx.x * 128;
    const bf16_t* Ag = wh   + (size_t)b * Tt * Ss + (size_t)tBase * Ss;
    const bf16_t* Bg = srch + (size_t)b * Ss * Dd + dBase;

    const int lane = threadIdx.x & 31;
    const int wave = threadIdx.x >> 5;
    const int wm = wave & 3, wn = wave >> 2;
    const int mr = lane & 15;
    const int kg = (lane >> 4) << 3;

    v8f zero = {0.f, 0.f, 0.f, 0.f, 0.f, 0.f, 0.f, 0.f};
    v8f acc[2][4];
#pragma unroll
    for (int i = 0; i < 2; ++i)
#pragma unroll
        for (int j = 0; j < 4; ++j) acc[i][j] = zero;

    stage_async_rowmajor(sA[0], Ag, Ss);
    stage_colmajor_bf(sB[0], Bg, Dd);
    wait_async0();
    __syncthreads();

    int p = 0;
    for (int k0 = 0; k0 < Ss; k0 += 32) {
        int kn = k0 + 32;
        if (kn < Ss) {
            stage_async_rowmajor(sA[p ^ 1], Ag + kn, Ss);
            stage_colmajor_bf(sB[p ^ 1], Bg + (size_t)kn * Dd, Dd);
        }
        v16bf af[2], bfm[4];
#pragma unroll
        for (int i = 0; i < 2; ++i) af[i] = frag_ld(&sA[p][wm * 32 + i * 16 + mr][kg]);
#pragma unroll
        for (int j = 0; j < 4; ++j) bfm[j] = frag_ld(&sB[p][wn * 64 + j * 16 + mr][kg]);
#pragma unroll
        for (int i = 0; i < 2; ++i)
#pragma unroll
            for (int j = 0; j < 4; ++j)
                acc[i][j] = __builtin_amdgcn_wmma_f32_16x16x32_bf16(
                    false, af[i], false, bfm[j], (short)0, acc[i][j], false, false);
        wait_async0();
        __syncthreads();
        p ^= 1;
    }

    bf16_t* ob = wsumh + (size_t)b * Tt * Dd;
    const int half = lane >> 4;
    const int col  = lane & 15;
#pragma unroll
    for (int i = 0; i < 2; ++i) {
        int mb = tBase + wm * 32 + i * 16 + half * 8;
#pragma unroll
        for (int j = 0; j < 4; ++j) {
            int n = dBase + wn * 64 + j * 16 + col;
#pragma unroll
            for (int r = 0; r < 8; ++r)
                ob[(size_t)(mb + r) * Dd + n] = f2bf(acc[i][j][r]);
        }
    }
}

// ---------------------------------------------------------------------------
// Kernel 4: out[r,o] = [wsum | tgt][r,:] . W[o,:] + bias[o]   (r = b*T + t)
// Single K=2048 GEMM, A source switches at k=1024. All-async staging.
// ---------------------------------------------------------------------------
__global__ __launch_bounds__(256) void k_out(const bf16_t* __restrict__ wsumh,
                                             const bf16_t* __restrict__ tgth,
                                             const bf16_t* __restrict__ Wh,
                                             const float*  __restrict__ bias,
                                             float*        __restrict__ out) {
    __shared__ __align__(16) bf16_t sA[2][128][LPAD];
    __shared__ __align__(16) bf16_t sB[2][128][LPAD];
    const int rBase = blockIdx.y * 128;   // global row in [0, B*T)
    const int oBase = blockIdx.x * 128;

    const int lane = threadIdx.x & 31;
    const int wave = threadIdx.x >> 5;
    const int wm = wave & 3, wn = wave >> 2;
    const int mr = lane & 15;
    const int kg = (lane >> 4) << 3;

    v8f zero = {0.f, 0.f, 0.f, 0.f, 0.f, 0.f, 0.f, 0.f};
    v8f acc[2][4];
#pragma unroll
    for (int i = 0; i < 2; ++i)
#pragma unroll
        for (int j = 0; j < 4; ++j) acc[i][j] = zero;

    const bf16_t* Bg = Wh + (size_t)oBase * (2 * Dd);
    auto aptr = [&](int k0) -> const bf16_t* {
        return (k0 < Dd) ? (wsumh + (size_t)rBase * Dd + k0)
                         : (tgth  + (size_t)rBase * Dd + (k0 - Dd));
    };

    stage_async_rowmajor(sA[0], aptr(0), Dd);
    stage_async_rowmajor(sB[0], Bg, 2 * Dd);
    wait_async0();
    __syncthreads();

    int p = 0;
    for (int k0 = 0; k0 < 2 * Dd; k0 += 32) {
        int kn = k0 + 32;
        if (kn < 2 * Dd) {
            stage_async_rowmajor(sA[p ^ 1], aptr(kn), Dd);
            stage_async_rowmajor(sB[p ^ 1], Bg + kn, 2 * Dd);
        }
        v16bf af[2], bfm[4];
#pragma unroll
        for (int i = 0; i < 2; ++i) af[i] = frag_ld(&sA[p][wm * 32 + i * 16 + mr][kg]);
#pragma unroll
        for (int j = 0; j < 4; ++j) bfm[j] = frag_ld(&sB[p][wn * 64 + j * 16 + mr][kg]);
#pragma unroll
        for (int i = 0; i < 2; ++i)
#pragma unroll
            for (int j = 0; j < 4; ++j)
                acc[i][j] = __builtin_amdgcn_wmma_f32_16x16x32_bf16(
                    false, af[i], false, bfm[j], (short)0, acc[i][j], false, false);
        wait_async0();
        __syncthreads();
        p ^= 1;
    }

    const int half = lane >> 4;
    const int col  = lane & 15;
#pragma unroll
    for (int i = 0; i < 2; ++i) {
        int mb = rBase + wm * 32 + i * 16 + half * 8;
#pragma unroll
        for (int j = 0; j < 4; ++j) {
            int o = oBase + wn * 64 + j * 16 + col;
            float bv = bias[o];
#pragma unroll
            for (int r = 0; r < 8; ++r)
                out[(size_t)(mb + r) * Oo + o] = acc[i][j][r] + bv;
        }
    }
}

extern "C" void kernel_launch(void* const* d_in, const int* in_sizes, int n_in,
                              void* d_out, int out_size, void* d_ws, size_t ws_size,
                              hipStream_t stream) {
    (void)in_sizes; (void)n_in; (void)out_size; (void)ws_size;
    const float* src  = (const float*)d_in[0];   // [B,S,D]
    const float* tgt  = (const float*)d_in[1];   // [B,T,D]
    const int*   mask = (const int*)  d_in[2];   // [T,B,S]
    const float* W    = (const float*)d_in[3];   // [O,2D]
    const float* bias = (const float*)d_in[4];   // [O]

    float* out  = (float*)d_out;                       // [B,T,O]
    float* wout = out + (size_t)Bb * Tt * Oo;          // [B,T,S] weight output

    // Workspace layout (bf16 operands), all 16B aligned
    char* wsp = (char*)d_ws;
    bf16_t* srch  = (bf16_t*)wsp;  wsp += (size_t)Bb * Ss * Dd * 2;
    bf16_t* tgth  = (bf16_t*)wsp;  wsp += (size_t)Bb * Tt * Dd * 2;
    bf16_t* Wh    = (bf16_t*)wsp;  wsp += (size_t)Oo * 2 * Dd * 2;
    bf16_t* wh    = (bf16_t*)wsp;  wsp += (size_t)Bb * Tt * Ss * 2;
    bf16_t* wsumh = (bf16_t*)wsp;

    dim3 blk(256);
    const int nSrc4 = Bb * Ss * Dd / 4, nTgt4 = Bb * Tt * Dd / 4, nW4 = Oo * 2 * Dd / 4;
    k_cvt<<<dim3(nSrc4 / 256), blk, 0, stream>>>(src, srch, nSrc4);
    k_cvt<<<dim3(nTgt4 / 256), blk, 0, stream>>>(tgt, tgth, nTgt4);
    k_cvt<<<dim3(nW4   / 256), blk, 0, stream>>>(W,   Wh,   nW4);

    k_scores <<<dim3(Ss / 128, Tt / 128, Bb),    blk, 0, stream>>>(srch, tgth, mask, wout);
    k_softmax<<<dim3((Bb * Ss) / 256),           blk, 0, stream>>>(wout, wh);
    k_wsum   <<<dim3(Dd / 128, Tt / 128, Bb),    blk, 0, stream>>>(wh, srch, wsumh);
    k_out    <<<dim3(Oo / 128, (Bb * Tt) / 128), blk, 0, stream>>>(wsumh, tgth, Wh, bias, out);
}
